// HebbianLinear_27891517620866
// MI455X (gfx1250) — matrix operand order repaired and checked by reading, not compiled, not run on hardware
//
#include <hip/hip_runtime.h>
#include <hip/hip_bf16.h>
#include <math.h>

// ---------------------------------------------------------------------------
// Types for CDNA5 WMMA (wave32): 16x16x32 bf16 -> f32
// ---------------------------------------------------------------------------
typedef __attribute__((ext_vector_type(16))) __bf16 bf16x16;
typedef __attribute__((ext_vector_type(8)))  __bf16 bf16x8;
typedef __attribute__((ext_vector_type(8)))  float  f32x8;

#define IN_DIM   1024
#define OUT_DIM  1024
#define RATE_C   1e-3f
#define LDS_PITCH 80            // 64B of K data + 16B pad (16 distinct banks/frag)
#define LDS_TILE  (128 * LDS_PITCH)   // 10240 B per 128x32 bf16 tile

// ---------------------------------------------------------------------------
// CDNA5 async global->LDS copy (ASYNCcnt) via inline asm
// ---------------------------------------------------------------------------
__device__ __forceinline__ void async_load_b128(unsigned lds_byte_addr, const void* gptr) {
  asm volatile("global_load_async_to_lds_b128 %0, %1, off"
               :: "v"(lds_byte_addr), "v"(gptr)
               : "memory");
}
__device__ __forceinline__ void wait_async0() {
  asm volatile("s_wait_asynccnt 0x0" ::: "memory");
}

// ---------------------------------------------------------------------------
// Fragment loaders (ISA 7.12.2 layouts, wave32)
// A (16x32, MxK): lane m=(l&15), kh=(l>>4): V0..3=K[kh*8..+7], V4..7=K[16+kh*8..+7]
// B (32x16, KxN) from N-major source: lane n=(l&15), kh=(l>>4): V0..3=K[kh*8..+7], V4..7=K[16+kh*8..+7]
// ---------------------------------------------------------------------------
__device__ __forceinline__ bf16x16 load_frag_a(const __bf16* p, int ld) {
  const int lane = threadIdx.x & 31;
  const int m = lane & 15, kh = lane >> 4;
  const __bf16* q = p + (size_t)m * ld + kh * 8;
  bf16x8 lo = *(const bf16x8*)q;
  bf16x8 hi = *(const bf16x8*)(q + 16);
  return __builtin_shufflevector(lo, hi, 0,1,2,3,4,5,6,7,8,9,10,11,12,13,14,15);
}

__device__ __forceinline__ bf16x16 load_frag_b(const __bf16* p, int ld) {
  const int lane = threadIdx.x & 31;
  const int n = lane & 15, kh = lane >> 4;
  const __bf16* q = p + (size_t)n * ld + kh * 8;
  bf16x8 lo = *(const bf16x8*)q;
  bf16x8 hi = *(const bf16x8*)(q + 16);
  return __builtin_shufflevector(lo, hi, 0,1,2,3,4,5,6,7,8,9,10,11,12,13,14,15);
}

// Same fragments, sourced from an LDS tile with LDS_PITCH-byte rows
__device__ __forceinline__ bf16x16 lds_frag_a(const char* p) {
  const int lane = threadIdx.x & 31;
  const int m = lane & 15, kh = lane >> 4;
  const char* q = p + m * LDS_PITCH + kh * 16;
  bf16x8 lo = *(const bf16x8*)q;
  bf16x8 hi = *(const bf16x8*)(q + 32);
  return __builtin_shufflevector(lo, hi, 0,1,2,3,4,5,6,7,8,9,10,11,12,13,14,15);
}

__device__ __forceinline__ bf16x16 lds_frag_b(const char* p) {
  const int lane = threadIdx.x & 31;
  const int n = lane & 15, kh = lane >> 4;
  const char* q = p + n * LDS_PITCH + kh * 16;
  bf16x8 lo = *(const bf16x8*)q;
  bf16x8 hi = *(const bf16x8*)(q + 32);
  return __builtin_shufflevector(lo, hi, 0,1,2,3,4,5,6,7,8,9,10,11,12,13,14,15);
}

// C/D layout (32-bit 16x16): VGPR r: lanes 0-15 -> M=r, lanes 16-31 -> M=8+r
__device__ __forceinline__ void store_tile_f32(float* dst, int ld, const f32x8& c) {
  const int lane = threadIdx.x & 31;
  const int n = lane & 15, mh = lane >> 4;
#pragma unroll
  for (int r = 0; r < 8; ++r)
    dst[(size_t)(mh * 8 + r) * ld + n] = c[r];
}

// ---------------------------------------------------------------------------
// Elementwise f32 -> bf16 convert
// ---------------------------------------------------------------------------
__global__ void __launch_bounds__(256) convert_bf16(const float* __restrict__ src,
                                                    __bf16* __restrict__ dst, long n) {
  long i = (long)blockIdx.x * blockDim.x + threadIdx.x;
  long stride = (long)gridDim.x * blockDim.x;
  for (; i < n; i += stride) dst[i] = (__bf16)src[i];
}

// Tiled transpose-convert: X[B, IN] f32 -> XT[IN, B] bf16
__global__ void __launch_bounds__(256) transpose_bf16(const float* __restrict__ X,
                                                      __bf16* __restrict__ XT, int Bn) {
  __shared__ float tile[32][33];
  const int tx = threadIdx.x & 31;
  const int ty = threadIdx.x >> 5;  // 0..7
  const int c0 = blockIdx.x * 32;   // along IN
  const int r0 = blockIdx.y * 32;   // along B
#pragma unroll
  for (int j = 0; j < 4; ++j)
    tile[ty + 8 * j][tx] = X[(size_t)(r0 + ty + 8 * j) * IN_DIM + c0 + tx];
  __syncthreads();
#pragma unroll
  for (int j = 0; j < 4; ++j)
    XT[(size_t)(c0 + ty + 8 * j) * Bn + r0 + tx] = (__bf16)tile[tx][ty + 8 * j];
}

// ---------------------------------------------------------------------------
// GEMM1: U[b,o] = sum_k Xbf[b,k]*Wbf[o,k] + bias[o]   (bf16 WMMA, f32 acc)
// 128 thr = 4 waves (2x2); block tile 128x128; wave tile 64x64 (4x4 frags)
// ---------------------------------------------------------------------------
__global__ void __launch_bounds__(128) gemm1_wmma(const __bf16* __restrict__ Xbf,
                                                  const __bf16* __restrict__ Wbf,
                                                  const float* __restrict__ bias,
                                                  float* __restrict__ U) {
  const int wave = threadIdx.x >> 5;
  const int lane = threadIdx.x & 31;
  const int row0 = blockIdx.y * 128 + (wave >> 1) * 64;  // M (batch rows)
  const int col0 = blockIdx.x * 128 + (wave & 1) * 64;   // N (out units)

  f32x8 acc[4][4];
#pragma unroll
  for (int nf = 0; nf < 4; ++nf) {
    const float bv = bias[col0 + nf * 16 + (lane & 15)];
#pragma unroll
    for (int mf = 0; mf < 4; ++mf)
#pragma unroll
      for (int r = 0; r < 8; ++r) acc[mf][nf][r] = bv;
  }

  for (int k0 = 0; k0 < IN_DIM; k0 += 32) {
    bf16x16 a[4], b[4];
#pragma unroll
    for (int mf = 0; mf < 4; ++mf)
      a[mf] = load_frag_a(Xbf + (size_t)(row0 + mf * 16) * IN_DIM + k0, IN_DIM);
#pragma unroll
    for (int nf = 0; nf < 4; ++nf)
      b[nf] = load_frag_b(Wbf + (size_t)(col0 + nf * 16) * IN_DIM + k0, IN_DIM);
#pragma unroll
    for (int mf = 0; mf < 4; ++mf)
#pragma unroll
      for (int nf = 0; nf < 4; ++nf)
        acc[mf][nf] = __builtin_amdgcn_wmma_f32_16x16x32_bf16(
            false, a[mf], false, b[nf], (short)0, acc[mf][nf], false, false);
  }

#pragma unroll
  for (int mf = 0; mf < 4; ++mf)
#pragma unroll
    for (int nf = 0; nf < 4; ++nf)
      store_tile_f32(U + (size_t)(row0 + mf * 16) * OUT_DIM + col0 + nf * 16,
                     OUT_DIM, acc[mf][nf]);
}

// ---------------------------------------------------------------------------
// Row stats: per batch row -> max, argmax (first-index ties), sum(exp(u-max))
// ---------------------------------------------------------------------------
__global__ void __launch_bounds__(256) row_stats(const float* __restrict__ U,
                                                 float* __restrict__ rowmax,
                                                 float* __restrict__ rowsum,
                                                 int* __restrict__ rowarg) {
  const int b = blockIdx.x;
  const int t = threadIdx.x;
  const float* row = U + (size_t)b * OUT_DIM;

  float vals[4];
  float mx = -INFINITY;
  int mi = 0;
#pragma unroll
  for (int j = 0; j < 4; ++j) {
    const int o = t + 256 * j;
    const float v = row[o];
    vals[j] = v;
    if (v > mx) { mx = v; mi = o; }
  }

  __shared__ float smx[256];
  __shared__ int smi[256];
  smx[t] = mx; smi[t] = mi;
  __syncthreads();
  for (int s = 128; s > 0; s >>= 1) {
    if (t < s) {
      const float ov = smx[t + s]; const int oi = smi[t + s];
      if (ov > smx[t] || (ov == smx[t] && oi < smi[t])) { smx[t] = ov; smi[t] = oi; }
    }
    __syncthreads();
  }
  const float rmax = smx[0];
  const int rarg = smi[0];
  __syncthreads();

  float se = 0.f;
#pragma unroll
  for (int j = 0; j < 4; ++j) se += __expf(vals[j] - rmax);
  smx[t] = se;
  __syncthreads();
  for (int s = 128; s > 0; s >>= 1) {
    if (t < s) smx[t] += smx[t + s];
    __syncthreads();
  }
  if (t == 0) { rowmax[b] = rmax; rowsum[b] = smx[0]; rowarg[b] = rarg; }
}

// ---------------------------------------------------------------------------
// Column pass: yn = sign * softmax(u); write YnT[o,b] (bf16, K-contig for
// GEMM2 A-operand); accumulate ssum[o] += yn*u via atomics.
// ---------------------------------------------------------------------------
__global__ void __launch_bounds__(256) col_pass(const float* __restrict__ U,
                                                const float* __restrict__ rowmax,
                                                const float* __restrict__ rowsum,
                                                const int* __restrict__ rowarg,
                                                __bf16* __restrict__ YnT,
                                                float* __restrict__ ssum,
                                                int Bn, int bchunk) {
  const int o = blockIdx.x * 256 + threadIdx.x;
  const int b0 = blockIdx.y * bchunk;
  float acc = 0.f;
  for (int b = b0; b < b0 + bchunk; ++b) {
    const float u = U[(size_t)b * OUT_DIM + o];
    const float y = __expf(u - rowmax[b]) / rowsum[b];
    const float yn = (rowarg[b] == o) ? y : -y;
    acc += yn * u;
    YnT[(size_t)o * Bn + b] = (__bf16)yn;
  }
  atomicAdd(&ssum[o], acc);
}

// ---------------------------------------------------------------------------
// GEMM2 (split-K): S[o,i] += sum_{k chunk} YnT[o,k]*XT[i,k]
// LDS-staged, double-buffered via global_load_async_to_lds_b128 (ASYNCcnt).
// 128 thr = 4 waves (2x2); block tile 128x128; wave tile 64x64; K-step 32.
// ---------------------------------------------------------------------------
__device__ __forceinline__ void stage_tiles(const __bf16* Ag, const __bf16* Bg, int ld,
                                            char* ldsA, char* ldsB) {
  const int t = threadIdx.x;  // 0..127
#pragma unroll
  for (int j = 0; j < 4; ++j) {
    const int chunk = t + 128 * j;           // 0..511
    const int row = chunk >> 2, seg = chunk & 3;
    async_load_b128((unsigned)(size_t)(ldsA + row * LDS_PITCH + seg * 16),
                    Ag + (size_t)row * ld + seg * 8);
    async_load_b128((unsigned)(size_t)(ldsB + row * LDS_PITCH + seg * 16),
                    Bg + (size_t)row * ld + seg * 8);
  }
}

__global__ void __launch_bounds__(128) gemm2_wmma(const __bf16* __restrict__ YnT,
                                                  const __bf16* __restrict__ XT,
                                                  float* __restrict__ S,
                                                  int Bn, int klen) {
  __shared__ __align__(16) char smem[4 * LDS_TILE];  // [A0][A1][B0][B1]

  const int wave = threadIdx.x >> 5;
  const int lane = threadIdx.x & 31;
  const int rowB = blockIdx.y * 128;                 // block base (OUT)
  const int colB = blockIdx.x * 128;                 // block base (IN)
  const int wrow = (wave >> 1) * 64;
  const int wcol = (wave & 1) * 64;
  const int kbeg = blockIdx.z * klen;
  const int kend = kbeg + klen;

  f32x8 acc[4][4];
#pragma unroll
  for (int mf = 0; mf < 4; ++mf)
#pragma unroll
    for (int nf = 0; nf < 4; ++nf)
#pragma unroll
      for (int r = 0; r < 8; ++r) acc[mf][nf][r] = 0.f;

  stage_tiles(YnT + (size_t)rowB * Bn + kbeg, XT + (size_t)colB * Bn + kbeg, Bn,
              smem, smem + 2 * LDS_TILE);
  int buf = 0;
  for (int k0 = kbeg; k0 < kend; k0 += 32) {
    wait_async0();
    __syncthreads();   // all waves' async copies into `buf` complete; prev reads done
    if (k0 + 32 < kend)
      stage_tiles(YnT + (size_t)rowB * Bn + (k0 + 32), XT + (size_t)colB * Bn + (k0 + 32),
                  Bn,
                  smem + (buf ^ 1) * LDS_TILE,
                  smem + (2 + (buf ^ 1)) * LDS_TILE);

    const char* curA = smem + buf * LDS_TILE;
    const char* curB = smem + (2 + buf) * LDS_TILE;
    bf16x16 a[4], b[4];
#pragma unroll
    for (int mf = 0; mf < 4; ++mf)
      a[mf] = lds_frag_a(curA + (wrow + mf * 16) * LDS_PITCH);
#pragma unroll
    for (int nf = 0; nf < 4; ++nf)
      b[nf] = lds_frag_b(curB + (wcol + nf * 16) * LDS_PITCH);
#pragma unroll
    for (int mf = 0; mf < 4; ++mf)
#pragma unroll
      for (int nf = 0; nf < 4; ++nf)
        acc[mf][nf] = __builtin_amdgcn_wmma_f32_16x16x32_bf16(
            false, a[mf], false, b[nf], (short)0, acc[mf][nf], false, false);
    buf ^= 1;
  }

  const int n = lane & 15, mh = lane >> 4;
#pragma unroll
  for (int mf = 0; mf < 4; ++mf)
#pragma unroll
    for (int nf = 0; nf < 4; ++nf)
#pragma unroll
      for (int r = 0; r < 8; ++r) {
        const int o = rowB + wrow + mf * 16 + mh * 8 + r;
        const int i = colB + wcol + nf * 16 + n;
        atomicAdd(&S[(size_t)o * IN_DIM + i], acc[mf][nf][r]);
      }
}

// ---------------------------------------------------------------------------
// W row norms -> adaptive rate[o] = RATE * |1 - ||W_o||_2|^0.5
// ---------------------------------------------------------------------------
__global__ void __launch_bounds__(256) row_norm(const float* __restrict__ Wf,
                                                float* __restrict__ rate) {
  const int o = blockIdx.x;
  const int t = threadIdx.x;
  float ss = 0.f;
#pragma unroll
  for (int j = 0; j < 4; ++j) {
    const float w = Wf[(size_t)o * IN_DIM + t + 256 * j];
    ss += w * w;
  }
  __shared__ float sb[256];
  sb[t] = ss;
  __syncthreads();
  for (int s = 128; s > 0; s >>= 1) {
    if (t < s) sb[t] += sb[t + s];
    __syncthreads();
  }
  if (t == 0) {
    const float d = fabsf(1.f - sqrtf(sb[0]));
    rate[o] = RATE_C * sqrtf(d);   // p = 0.5
  }
}

// out[o,i] = rate[o] * (S[o,i]/B - (ssum[o]/B) * W[o,i])
__global__ void __launch_bounds__(256) finalize(const float* __restrict__ S,
                                                const float* __restrict__ Wf,
                                                const float* __restrict__ rate,
                                                const float* __restrict__ ssum,
                                                float* __restrict__ out, float invB) {
  const int i = blockIdx.x * 256 + threadIdx.x;
  const int o = blockIdx.y;
  const size_t idx = (size_t)o * IN_DIM + i;
  out[idx] = rate[o] * (S[idx] * invB - (ssum[o] * invB) * Wf[idx]);
}

__global__ void __launch_bounds__(256) zero_f32(float* __restrict__ p, long n) {
  long i = (long)blockIdx.x * blockDim.x + threadIdx.x;
  long stride = (long)gridDim.x * blockDim.x;
  for (; i < n; i += stride) p[i] = 0.f;
}

// ---------------------------------------------------------------------------
// Host launcher
// ---------------------------------------------------------------------------
extern "C" void kernel_launch(void* const* d_in, const int* in_sizes, int n_in,
                              void* d_out, int out_size, void* d_ws, size_t ws_size,
                              hipStream_t stream) {
  const float* x    = (const float*)d_in[0];   // [B, IN]
  const float* W    = (const float*)d_in[1];   // [OUT, IN]
  const float* bias = (const float*)d_in[2];   // [OUT]
  const int Bn = in_sizes[0] / IN_DIM;         // 8192

  char* ws = (char*)d_ws;
  size_t off = 0;
  auto carve = [&](size_t bytes) {
    void* p = ws + off;
    off += (bytes + 255) & ~(size_t)255;
    return p;
  };
  float*  U      = (float*) carve((size_t)Bn * OUT_DIM * 4);   // 32 MB
  __bf16* Xbf    = (__bf16*)carve((size_t)Bn * IN_DIM * 2);    // 16 MB
  __bf16* XT     = (__bf16*)carve((size_t)IN_DIM * Bn * 2);    // 16 MB
  __bf16* Wbf    = (__bf16*)carve((size_t)OUT_DIM * IN_DIM * 2);
  __bf16* YnT    = (__bf16*)carve((size_t)OUT_DIM * Bn * 2);   // 16 MB
  float*  S      = (float*) carve((size_t)OUT_DIM * IN_DIM * 4);
  float*  rowmax = (float*) carve((size_t)Bn * 4);
  float*  rowsum = (float*) carve((size_t)Bn * 4);
  int*    rowarg = (int*)   carve((size_t)Bn * 4);
  float*  ssum   = (float*) carve((size_t)OUT_DIM * 4);
  float*  rate   = (float*) carve((size_t)OUT_DIM * 4);

  // Stage 0: precision conversion + transpose + independent small reductions
  convert_bf16<<<4096, 256, 0, stream>>>(x, Xbf, (long)Bn * IN_DIM);
  convert_bf16<<<1024, 256, 0, stream>>>(W, Wbf, (long)OUT_DIM * IN_DIM);
  transpose_bf16<<<dim3(IN_DIM / 32, Bn / 32), 256, 0, stream>>>(x, XT, Bn);
  row_norm<<<OUT_DIM, 256, 0, stream>>>(W, rate);
  zero_f32<<<1024, 256, 0, stream>>>(S, (long)OUT_DIM * IN_DIM);
  zero_f32<<<4, 256, 0, stream>>>(ssum, (long)OUT_DIM);

  // Stage 1: u = x @ W^T + b  (WMMA bf16, 64x64 wave tiles, direct loads)
  gemm1_wmma<<<dim3(OUT_DIM / 128, Bn / 128), 128, 0, stream>>>(Xbf, Wbf, bias, U);

  // Stage 2: softmax stats + yn / sum(yn*u)
  row_stats<<<Bn, 256, 0, stream>>>(U, rowmax, rowsum, rowarg);
  col_pass<<<dim3(OUT_DIM / 256, Bn / 128), 256, 0, stream>>>(
      U, rowmax, rowsum, rowarg, YnT, ssum, Bn, 128);

  // Stage 3: S = yn^T @ x  (WMMA bf16, async-LDS double buffered, split-K=8)
  const int ksplit = 8;
  gemm2_wmma<<<dim3(IN_DIM / 128, OUT_DIM / 128, ksplit), 128, 0, stream>>>(
      YnT, XT, S, Bn, Bn / ksplit);

  // Stage 4: adaptive-rate scaling
  finalize<<<dim3(IN_DIM / 256, OUT_DIM), 256, 0, stream>>>(
      S, W, rate, ssum, (float*)d_out, 1.0f / (float)Bn);
}
